// GAT_Net_1_81243601371614
// MI455X (gfx1250) — matrix operand order repaired and checked by reading, not compiled, not run on hardware
//
#include <hip/hip_runtime.h>
#include <hip/hip_bf16.h>

#define N_NODES 100000
#define N_EDGES 1600000
#define IN_CH   128
#define HIDDEN  48
#define OUT_CH  16
#define NEG_SLOPE 0.2f
#define TOTAL_E (N_EDGES + N_NODES)

typedef __attribute__((ext_vector_type(16))) _Float16 v16h;
typedef __attribute__((ext_vector_type(8)))  float    v8f;

// ---------------------------------------------------------------------------
// Kernel 0: pack W (row-major f32 [128 x 48]) into per-lane WMMA B fragments
// as f16: pB[(kchunk*3 + ntile)*32 + lane][16]. 12 KB total, L2 resident.
// B k order used here (contiguous, same as A; not AMD's table order): lanes 0-15 hold K=k0+0..15, lanes 16-31 K=k0+16..31,
// column n = lane & 15; element i of the fragment is K=kb+i.
// ---------------------------------------------------------------------------
__global__ void __launch_bounds__(128)
gat_pack_B(const float* __restrict__ W, _Float16* __restrict__ pB) {
  const int t = blockIdx.x * blockDim.x + threadIdx.x;   // 0..383
  if (t >= 4 * 3 * 32) return;
  const int lane = t & 31;
  const int nt   = (t >> 5) % 3;
  const int kc   = t / 96;
  const int half = lane >> 4;
  const int l15  = lane & 15;
  const int kb   = kc * 32 + half * 16;
  const int n    = nt * 16 + l15;
  _Float16* dst = pB + ((size_t)t) * 16;
  #pragma unroll
  for (int i = 0; i < 16; ++i) dst[i] = (_Float16)W[(kb + i) * HIDDEN + n];
}

// ---------------------------------------------------------------------------
// Kernel 1: h = x @ W via v_wmma_f32_16x16x32_f16. One wave per 16-row tile.
// M = 6250*16 exact, N = 3*16 exact, K = 4*32 exact -> no tails.
// ---------------------------------------------------------------------------
__global__ void __launch_bounds__(256)
gat_gemm_h(const float* __restrict__ x, const v16h* __restrict__ pB,
           float* __restrict__ h) {
  const int wave = (blockIdx.x * blockDim.x + threadIdx.x) >> 5;
  const int lane = threadIdx.x & 31;
  const int m0   = wave * 16;
  if (m0 >= N_NODES) return;              // uniform per wave: EXEC all-1 for WMMA

  const int half = lane >> 4;             // 0: lanes 0-15, 1: lanes 16-31
  const int l15  = lane & 15;
  const float* xrow = x + (size_t)(m0 + l15) * IN_CH;

  v8f acc0 = {}, acc1 = {}, acc2 = {};

  #pragma unroll
  for (int kc = 0; kc < 4; ++kc) {
    const int k0 = kc * 32;
    // A (16x32 f16): half 0 holds K=k0+0..15, half 1 holds K=k0+16..31
    //                (element i is K=ka+i, matching packed B).
    v16h a;
    const int ka = k0 + half * 16;
    #pragma unroll
    for (int i = 0; i < 8; ++i) a[i]     = (_Float16)xrow[ka + i];
    #pragma unroll
    for (int i = 0; i < 8; ++i) a[8 + i] = (_Float16)xrow[ka + 8 + i];

    // B fragments: one 32-byte vector load per (kc, ntile).
    const v16h b0 = pB[(kc * 3 + 0) * 32 + lane];
    const v16h b1 = pB[(kc * 3 + 1) * 32 + lane];
    const v16h b2 = pB[(kc * 3 + 2) * 32 + lane];

    acc0 = __builtin_amdgcn_wmma_f32_16x16x32_f16(false, a, false, b0,
                                                  (short)0, acc0, false, false);
    acc1 = __builtin_amdgcn_wmma_f32_16x16x32_f16(false, a, false, b1,
                                                  (short)0, acc1, false, false);
    acc2 = __builtin_amdgcn_wmma_f32_16x16x32_f16(false, a, false, b2,
                                                  (short)0, acc2, false, false);
  }

  // D layout: VGPR r -> row m0 + r + 8*half, col = nt*16 + l15
  #pragma unroll
  for (int nt = 0; nt < 3; ++nt) {
    const v8f acc = (nt == 0) ? acc0 : (nt == 1) ? acc1 : acc2;
    float* hb = h + (size_t)(m0 + half * 8) * HIDDEN + nt * 16 + l15;
    #pragma unroll
    for (int r = 0; r < 8; ++r) hb[(size_t)r * HIDDEN] = acc[r];
  }
}

// ---------------------------------------------------------------------------
// Kernel 2: per-node attention dots a_src[n] = h[n,:]·att_src, same for dst.
// ---------------------------------------------------------------------------
__global__ void __launch_bounds__(256)
gat_adots(const float* __restrict__ h, const float* __restrict__ att_src,
          const float* __restrict__ att_dst, float* __restrict__ a_src,
          float* __restrict__ a_dst) {
  const int n = blockIdx.x * blockDim.x + threadIdx.x;
  if (n >= N_NODES) return;
  const float* hr = h + (size_t)n * HIDDEN;
  float as = 0.f, ad = 0.f;
  #pragma unroll
  for (int c = 0; c < HIDDEN; ++c) {
    const float v = hr[c];
    as = fmaf(v, att_src[c], as);
    ad = fmaf(v, att_dst[c], ad);
  }
  a_src[n] = as;
  a_dst[n] = ad;
}

// Order-preserving float -> u32 mapping (monotonic), for atomic max.
__device__ __forceinline__ unsigned f2ord(float f) {
  const int fi = __float_as_int(f);
  return (unsigned)(fi ^ ((fi >> 31) | 0x80000000));
}
__device__ __forceinline__ float ord2f(unsigned u) {
  const int fi = (u & 0x80000000u) ? (int)(u ^ 0x80000000u) : (int)~u;
  return __int_as_float(fi);
}

// ---------------------------------------------------------------------------
// Kernel 3: edge logits (leaky relu) + segment max via atomicMax on mapped u32.
// Edges e < N_EDGES come from edge_index; e >= N_EDGES are self loops.
// ---------------------------------------------------------------------------
__global__ void __launch_bounds__(256)
gat_edge_logits(const int* __restrict__ ei, const float* __restrict__ a_src,
                const float* __restrict__ a_dst, float* __restrict__ logits,
                unsigned* __restrict__ mu) {
  const int e = blockIdx.x * blockDim.x + threadIdx.x;
  if (e >= TOTAL_E) return;
  int s, d;
  if (e < N_EDGES) { s = ei[e]; d = ei[N_EDGES + e]; }
  else             { s = d = e - N_EDGES; }
  float l = a_src[s] + a_dst[d];
  l = (l > 0.f) ? l : NEG_SLOPE * l;
  logits[e] = l;
  atomicMax(&mu[d], f2ord(l));
}

// ---------------------------------------------------------------------------
// Kernel 4: e = exp(logit - m[dst]); segment-sum denom via atomicAdd.
// Overwrites logits buffer with exp values.
// ---------------------------------------------------------------------------
__global__ void __launch_bounds__(256)
gat_edge_exp(const int* __restrict__ ei, float* __restrict__ logits,
             const unsigned* __restrict__ mu, float* __restrict__ denom) {
  const int e = blockIdx.x * blockDim.x + threadIdx.x;
  if (e >= TOTAL_E) return;
  int d;
  if (e < N_EDGES) d = ei[N_EDGES + e];
  else             d = e - N_EDGES;
  const float ex = __expf(logits[e] - ord2f(mu[d]));
  logits[e] = ex;
  atomicAdd(&denom[d], ex);
}

// ---------------------------------------------------------------------------
// Kernel 5: scatter agg[dst] += h[src] * alpha. 16 lanes per edge, 3 ch/lane.
// h and agg (19.2 MB each) are L2-resident -> random traffic stays on-chip.
// ---------------------------------------------------------------------------
__global__ void __launch_bounds__(256)
gat_scatter(const int* __restrict__ ei, const float* __restrict__ h,
            const float* __restrict__ exs, const float* __restrict__ denom,
            float* __restrict__ agg) {
  const long long t = (long long)blockIdx.x * blockDim.x + threadIdx.x;
  const int e  = (int)(t >> 4);
  const int c0 = (int)(t & 15);
  if (e >= TOTAL_E) return;
  int s, d;
  if (e < N_EDGES) { s = ei[e]; d = ei[N_EDGES + e]; }
  else             { s = d = e - N_EDGES; }
  const float alpha = exs[e] / denom[d];
  const float* hr = h   + (size_t)s * HIDDEN;
  float*       ag = agg + (size_t)d * HIDDEN;
  atomicAdd(&ag[c0],      hr[c0]      * alpha);
  atomicAdd(&ag[c0 + 16], hr[c0 + 16] * alpha);
  atomicAdd(&ag[c0 + 32], hr[c0 + 32] * alpha);
}

// ---------------------------------------------------------------------------
// Kernel 6: out = log_softmax(elu(agg + gat_bias) @ lin_W + lin_b).
// lin_W (48x16), gat_bias, lin_b staged in LDS.
// ---------------------------------------------------------------------------
__global__ void __launch_bounds__(256)
gat_final(const float* __restrict__ agg, const float* __restrict__ gat_bias,
          const float* __restrict__ lin_W, const float* __restrict__ lin_b,
          float* __restrict__ out) {
  __shared__ float sW[HIDDEN * OUT_CH];
  __shared__ float sb[HIDDEN];
  __shared__ float slb[OUT_CH];
  for (int i = threadIdx.x; i < HIDDEN * OUT_CH; i += blockDim.x) sW[i] = lin_W[i];
  if (threadIdx.x < HIDDEN) sb[threadIdx.x]  = gat_bias[threadIdx.x];
  if (threadIdx.x < OUT_CH) slb[threadIdx.x] = lin_b[threadIdx.x];
  __syncthreads();

  const int node = blockIdx.x * blockDim.x + threadIdx.x;
  if (node >= N_NODES) return;

  float acc[OUT_CH];
  #pragma unroll
  for (int j = 0; j < OUT_CH; ++j) acc[j] = slb[j];

  const float* ar = agg + (size_t)node * HIDDEN;
  #pragma unroll 4
  for (int k = 0; k < HIDDEN; ++k) {
    float v = ar[k] + sb[k];
    v = (v > 0.f) ? v : (__expf(v) - 1.f);           // ELU
    const float* wr = &sW[k * OUT_CH];
    #pragma unroll
    for (int j = 0; j < OUT_CH; ++j) acc[j] = fmaf(v, wr[j], acc[j]);
  }

  float mx = acc[0];
  #pragma unroll
  for (int j = 1; j < OUT_CH; ++j) mx = fmaxf(mx, acc[j]);
  float ssum = 0.f;
  #pragma unroll
  for (int j = 0; j < OUT_CH; ++j) ssum += __expf(acc[j] - mx);
  const float lse = __logf(ssum) + mx;

  float* orow = out + (size_t)node * OUT_CH;
  #pragma unroll
  for (int j = 0; j < OUT_CH; ++j) orow[j] = acc[j] - lse;
}

// ---------------------------------------------------------------------------
static inline size_t align256(size_t x) { return (x + 255) & ~(size_t)255; }

extern "C" void kernel_launch(void* const* d_in, const int* in_sizes, int n_in,
                              void* d_out, int out_size, void* d_ws, size_t ws_size,
                              hipStream_t stream) {
  const float* x        = (const float*)d_in[0];
  const int*   ei       = (const int*)  d_in[1];   // [2, N_EDGES] flat
  const float* W        = (const float*)d_in[2];
  const float* att_src  = (const float*)d_in[3];
  const float* att_dst  = (const float*)d_in[4];
  const float* gat_bias = (const float*)d_in[5];
  const float* lin_W    = (const float*)d_in[6];
  const float* lin_b    = (const float*)d_in[7];
  float* out = (float*)d_out;
  (void)in_sizes; (void)n_in; (void)out_size; (void)ws_size;

  // Workspace carve-up
  char* ws = (char*)d_ws;
  float*    h      = (float*)ws;     ws += align256((size_t)N_NODES * HIDDEN * 4);
  float*    a_src  = (float*)ws;     ws += align256((size_t)N_NODES * 4);
  float*    a_dst  = (float*)ws;     ws += align256((size_t)N_NODES * 4);
  unsigned* mu     = (unsigned*)ws;  ws += align256((size_t)N_NODES * 4);
  float*    denom  = (float*)ws;     ws += align256((size_t)N_NODES * 4);
  float*    logits = (float*)ws;     ws += align256((size_t)TOTAL_E * 4);
  float*    agg    = (float*)ws;     ws += align256((size_t)N_NODES * HIDDEN * 4);
  _Float16* pB     = (_Float16*)ws;  ws += align256((size_t)4 * 3 * 32 * 16 * 2);

  // Zero-init atomic targets (mapped-order 0 == smaller than any real float)
  hipMemsetAsync(mu,    0, (size_t)N_NODES * 4, stream);
  hipMemsetAsync(denom, 0, (size_t)N_NODES * 4, stream);
  hipMemsetAsync(agg,   0, (size_t)N_NODES * HIDDEN * 4, stream);

  // 0) pack W into f16 WMMA B fragments (12 KB)
  gat_pack_B<<<3, 128, 0, stream>>>(W, pB);
  // 1) h = x @ W (WMMA): 6250 waves of 16 rows, 8 waves per block
  {
    const int waves = N_NODES / 16;                // 6250
    const int blocks = (waves + 7) / 8;            // 782
    gat_gemm_h<<<blocks, 256, 0, stream>>>(x, (const v16h*)pB, h);
  }
  // 2) attention dots
  gat_adots<<<(N_NODES + 255) / 256, 256, 0, stream>>>(h, att_src, att_dst,
                                                       a_src, a_dst);
  // 3) logits + segment max
  gat_edge_logits<<<(TOTAL_E + 255) / 256, 256, 0, stream>>>(ei, a_src, a_dst,
                                                             logits, mu);
  // 4) exp + segment sum
  gat_edge_exp<<<(TOTAL_E + 255) / 256, 256, 0, stream>>>(ei, logits, mu, denom);
  // 5) weighted scatter (16 lanes per edge)
  {
    const long long threads = (long long)TOTAL_E * 16;
    const int blocks = (int)((threads + 255) / 256);
    gat_scatter<<<blocks, 256, 0, stream>>>(ei, h, logits, denom, agg);
  }
  // 6) ELU + linear + log_softmax
  gat_final<<<(N_NODES + 255) / 256, 256, 0, stream>>>(agg, gat_bias, lin_W,
                                                       lin_b, out);
}
